// WanSelfAttention_11278584119435
// MI455X (gfx1250) — matrix operand order repaired and checked by reading, not compiled, not run on hardware
//
#include <hip/hip_runtime.h>

typedef __attribute__((ext_vector_type(16))) _Float16 v16h;
typedef __attribute__((ext_vector_type(8)))  _Float16 v8h;
typedef __attribute__((ext_vector_type(8)))  float    v8f;
typedef __attribute__((address_space(3)))    _Float16 lds_f16_t;

#define LSEQ 5000
#define DIM  1536
#define NH   12
#define HD   128
#define LPAD 5120

#define WMMA_F16(a, b, c) \
  __builtin_amdgcn_wmma_f32_16x16x32_f16(false, (a), false, (b), (short)0, (c), false, false)

// Build a 16-half WMMA operand from two 8-half chunks.
static __device__ __forceinline__ v16h ld_frag2(const _Float16* p0, const _Float16* p1) {
  v16h r;
  *reinterpret_cast<v8h*>(&r)       = *reinterpret_cast<const v8h*>(p0);
  *(reinterpret_cast<v8h*>(&r) + 1) = *reinterpret_cast<const v8h*>(p1);
  return r;
}

// Async DMA: 16B per lane, global -> LDS, tracked by ASYNCcnt (no VGPR round-trip).
static __device__ __forceinline__ void async_b128(const _Float16* g, _Float16* l) {
  unsigned loff = (unsigned)(unsigned long long)(lds_f16_t*)l;  // LDS byte offset
  unsigned long long ga = (unsigned long long)(size_t)g;
  asm volatile("global_load_async_to_lds_b128 %0, %1, off"
               :: "v"(loff), "v"(ga) : "memory");
}

// ---------------------------------------------------------------------------
// f32 -> f16 cast with row padding (rows >= srcRows zero-filled)
// ---------------------------------------------------------------------------
__global__ void cast_pad_kernel(const float* __restrict__ src, _Float16* __restrict__ dst,
                                int srcRows, int cols, int dstRows) {
  long total = (long)dstRows * cols;
  for (long i = (long)blockIdx.x * blockDim.x + threadIdx.x; i < total;
       i += (long)gridDim.x * blockDim.x) {
    long r = i / cols;
    dst[i] = (r < srcRows) ? (_Float16)src[i] : (_Float16)0.0f;
  }
}

// ---------------------------------------------------------------------------
// f32 W[K][N] -> f16 Wt[N][K] (transpose at cast time so GEMM B panels are
// row-major and can be staged with async B128 copies).
// ---------------------------------------------------------------------------
__global__ void cast_transpose_kernel(const float* __restrict__ src, _Float16* __restrict__ dst,
                                      int K, int N) {
  long total = (long)K * N;
  for (long i = (long)blockIdx.x * blockDim.x + threadIdx.x; i < total;
       i += (long)gridDim.x * blockDim.x) {
    long n = i / K, k = i % K;
    dst[i] = (_Float16)src[k * N + n];
  }
}

// ---------------------------------------------------------------------------
// WMMA GEMM: C[M,N] = A[M,K](f16) * Bt[N,K](f16)^T + bias, out f16 or f32.
// WG tile 128x128, 8 waves in 2(M) x 4(N), each wave 64x32 = 8 WMMA accs.
// Tiles staged via global_load_async_to_lds_b128, double-buffered.
// ---------------------------------------------------------------------------
__global__ void __launch_bounds__(256)
gemm_f16_kernel(const _Float16* __restrict__ A, const _Float16* __restrict__ Bt,
                const float* __restrict__ bias, _Float16* __restrict__ outH,
                float* __restrict__ outF, int M, int N, int K, int Mlimit) {
  __shared__ _Float16 As[2][128 * 40];   // M x 32, stride 40 halves (16B-aligned rows)
  __shared__ _Float16 Bs[2][128 * 40];   // N x 32, stride 40 halves
  const int tid  = threadIdx.x;
  const int lane = tid & 31, wv = tid >> 5;
  const int lm   = lane & 15, hb = lane >> 4;
  const int koff = hb * 16;
  const int mbase = blockIdx.x * 128, nbase = blockIdx.y * 128;
  const int Moff = (wv >> 2) * 64, Noff = (wv & 3) * 32;

  v8f acc[4][2];
#pragma unroll
  for (int mi = 0; mi < 4; ++mi)
#pragma unroll
    for (int ni = 0; ni < 2; ++ni)
#pragma unroll
      for (int i = 0; i < 8; ++i) acc[mi][ni][i] = 0.0f;

  // Issue one 128x32 A panel + 128x32 B panel into LDS buffer `buf`.
  // 4 async instructions per wave per tile.
  auto issue_tile = [&](int k0, int buf) {
#pragma unroll
    for (int it = 0; it < 2; ++it) {
      int c = tid + it * 256;
      int row = c >> 2, off = (c & 3) * 8;
      async_b128(A  + (size_t)(mbase + row) * K + k0 + off, &As[buf][row * 40 + off]);
      async_b128(Bt + (size_t)(nbase + row) * K + k0 + off, &Bs[buf][row * 40 + off]);
    }
  };

  const int nk = K >> 5;
  issue_tile(0, 0);
  for (int kt = 0; kt < nk; ++kt) {
    const int cur = kt & 1;
    if (kt + 1 < nk) {
      issue_tile((kt + 1) << 5, cur ^ 1);
      // in-order retirement: <=4 outstanding => tile kt's 4 copies landed
      asm volatile("s_wait_asynccnt 0x4" ::: "memory");
    } else {
      asm volatile("s_wait_asynccnt 0x0" ::: "memory");
    }
    __syncthreads();   // RAW: everyone's tile-kt copies visible

    v16h af[4], bf[2];
#pragma unroll
    for (int mi = 0; mi < 4; ++mi) {
      const _Float16* p = &As[cur][(Moff + mi * 16 + lm) * 40 + koff];
      af[mi] = ld_frag2(p, p + 8);          // A layout: contiguous 16 per half
    }
#pragma unroll
    for (int ni = 0; ni < 2; ++ni) {
      const _Float16* p = &Bs[cur][(Noff + ni * 16 + lm) * 40 + hb * 16];
      bf[ni] = ld_frag2(p, p + 8);          // B layout: contiguous 16 per half
    }
#pragma unroll
    for (int mi = 0; mi < 4; ++mi)
#pragma unroll
      for (int ni = 0; ni < 2; ++ni)
        acc[mi][ni] = WMMA_F16(af[mi], bf[ni], acc[mi][ni]);
    __syncthreads();   // WAR: done reading buf `cur` before it is refilled
  }

#pragma unroll
  for (int mi = 0; mi < 4; ++mi) {
#pragma unroll
    for (int ni = 0; ni < 2; ++ni) {
      int col = nbase + Noff + ni * 16 + lm;
      float bv = bias ? bias[col] : 0.0f;
#pragma unroll
      for (int i = 0; i < 8; ++i) {
        int row = mbase + Moff + mi * 16 + i + hb * 8;
        if (row < Mlimit) {
          float v = acc[mi][ni][i] + bv;
          if (outH) outH[(size_t)row * N + col] = (_Float16)v;
          else      outF[(size_t)row * N + col] = v;
        }
      }
    }
  }
}

// ---------------------------------------------------------------------------
// RMSNorm (over 1536) + RoPE, output head-major [h][Lp][128] f16, pad zeroed.
// d=128 -> c=64 pairs, split c0=22 (f), c1=21 (h), c1=21 (w).
// ---------------------------------------------------------------------------
__global__ void __launch_bounds__(256)
rmsnorm_rope_kernel(const _Float16* __restrict__ src, const float* __restrict__ g,
                    const float* __restrict__ freqs, const int* __restrict__ grid_sizes,
                    _Float16* __restrict__ dst, int L, int Lp) {
  const int r = blockIdx.x, t = threadIdx.x;
  if (r >= L) {
    for (int e = t; e < DIM; e += 256) {
      int head = e >> 7;
      dst[(size_t)head * Lp * HD + (size_t)r * HD + (e & 127)] = (_Float16)0.0f;
    }
    return;
  }
  __shared__ float red[256];
  float ss = 0.f;
  for (int e = t; e < DIM; e += 256) {
    float v = (float)src[(size_t)r * DIM + e];
    ss += v * v;
  }
  red[t] = ss;
  __syncthreads();
  for (int s = 128; s > 0; s >>= 1) {
    if (t < s) red[t] += red[t + s];
    __syncthreads();
  }
  float inv = rsqrtf(red[0] / (float)DIM + 1e-6f);

  int gf = grid_sizes[0], gh = grid_sizes[1], gw = grid_sizes[2];
  int sl = gf * gh * gw;
  if (r < sl) {
    int hw = gh * gw;
    int fi = r / hw, rem = r % hw;
    int hi2 = rem / gw, wi = rem % gw;
    for (int p = t; p < NH * 64; p += 256) {
      int head = p >> 6, j = p & 63;
      int pos = (j < 22) ? fi : ((j < 43) ? hi2 : wi);
      float ang = freqs[pos * 64 + j];
      float cs = __cosf(ang), sn = __sinf(ang);
      int e0 = head * HD + 2 * j;
      float re = (float)src[(size_t)r * DIM + e0]     * inv * g[e0];
      float im = (float)src[(size_t)r * DIM + e0 + 1] * inv * g[e0 + 1];
      size_t ob = (size_t)head * Lp * HD + (size_t)r * HD + 2 * j;
      dst[ob]     = (_Float16)(re * cs - im * sn);
      dst[ob + 1] = (_Float16)(re * sn + im * cs);
    }
  } else {
    for (int e = t; e < DIM; e += 256) {
      int head = e >> 7;
      dst[(size_t)head * Lp * HD + (size_t)r * HD + (e & 127)] =
          (_Float16)((float)src[(size_t)r * DIM + e] * inv * g[e]);
    }
  }
}

// ---------------------------------------------------------------------------
// Pack V transposed per head: [h][d][Lp] f16 (pad rows zero).
// ---------------------------------------------------------------------------
__global__ void vpack_kernel(const _Float16* __restrict__ src, _Float16* __restrict__ dst,
                             int L, int Lp) {
  int r = blockIdx.x;
  for (int e = threadIdx.x; e < DIM; e += blockDim.x) {
    int head = e >> 7, dd = e & 127;
    _Float16 v = (r < L) ? src[(size_t)r * DIM + e] : (_Float16)0.0f;
    dst[(size_t)head * HD * Lp + (size_t)dd * Lp + r] = v;
  }
}

// ---------------------------------------------------------------------------
// Flash attention. Grid: (Lp/128, NH). 8 waves/WG, each wave owns 16 q rows.
// Q [h][Lp][128], K [h][Lp][128], V transposed [h][128][Lp]. Out f16 [Lp][1536].
// ---------------------------------------------------------------------------
__global__ void __launch_bounds__(256)
attn_kernel(const _Float16* __restrict__ Q, const _Float16* __restrict__ K,
            const _Float16* __restrict__ V, _Float16* __restrict__ O,
            const int* __restrict__ seq_lens, int Lp) {
  __shared__ _Float16 Psh[8][16 * 40];   // per-wave P transpose scratch
  const int h = blockIdx.y;
  const int Lvalid = seq_lens[0];
  const int tid = threadIdx.x;
  const int wave = tid >> 5, lane = tid & 31;
  const int lm = lane & 15, hb = lane >> 4;
  const int koff = hb * 16;
  const int q0 = blockIdx.x * 128 + wave * 16;
  const _Float16* Qh = Q + (size_t)h * Lp * HD;
  const _Float16* Kh = K + (size_t)h * Lp * HD;
  const _Float16* Vh = V + (size_t)h * HD * Lp;

  v16h qf[4];
#pragma unroll
  for (int kk = 0; kk < 4; ++kk) {
    const _Float16* p = Qh + (size_t)(q0 + lm) * HD + kk * 32 + koff;
    qf[kk] = ld_frag2(p, p + 8);
  }
  const float scale = 0.08838834764831845f;  // 1/sqrt(128)

  float mrun[8], lrun[8];
  v8f oacc[8];
#pragma unroll
  for (int i = 0; i < 8; ++i) { mrun[i] = -1e30f; lrun[i] = 0.f; }
#pragma unroll
  for (int gI = 0; gI < 8; ++gI)
#pragma unroll
    for (int i = 0; i < 8; ++i) oacc[gI][i] = 0.f;

  for (int j0 = 0; j0 < Lp; j0 += 32) {
    // ---- S = Q * K^T for a 16x32 key tile ----
    v8f s0, s1;
#pragma unroll
    for (int i = 0; i < 8; ++i) { s0[i] = 0.f; s1[i] = 0.f; }
#pragma unroll
    for (int kk = 0; kk < 4; ++kk) {
      const _Float16* p0 = Kh + (size_t)(j0 + lm) * HD + kk * 32 + hb * 16;
      const _Float16* p1 = Kh + (size_t)(j0 + 16 + lm) * HD + kk * 32 + hb * 16;
      v16h b0 = ld_frag2(p0, p0 + 8);
      v16h b1 = ld_frag2(p1, p1 + 8);
      s0 = WMMA_F16(qf[kk], b0, s0);
      s1 = WMMA_F16(qf[kk], b1, s1);
    }
    // ---- scale + key mask (column index depends only on lane) ----
    const bool v0 = (j0 + lm) < Lvalid;
    const bool v1 = (j0 + 16 + lm) < Lvalid;
#pragma unroll
    for (int i = 0; i < 8; ++i) {
      s0[i] = v0 ? s0[i] * scale : -1e30f;
      s1[i] = v1 ? s1[i] * scale : -1e30f;
    }
    // ---- online softmax, per C-layout row slot i (rows i and i+8) ----
#pragma unroll
    for (int i = 0; i < 8; ++i) {
      float mx = fmaxf(s0[i], s1[i]);
      mx = fmaxf(mx, __shfl_xor(mx, 8, 16));
      mx = fmaxf(mx, __shfl_xor(mx, 4, 16));
      mx = fmaxf(mx, __shfl_xor(mx, 2, 16));
      mx = fmaxf(mx, __shfl_xor(mx, 1, 16));
      float mnew = fmaxf(mrun[i], mx);
      float corr = __expf(mrun[i] - mnew);
      float p0v = __expf(s0[i] - mnew);
      float p1v = __expf(s1[i] - mnew);
      s0[i] = p0v; s1[i] = p1v;
      float rs = p0v + p1v;
      rs += __shfl_xor(rs, 8, 16);
      rs += __shfl_xor(rs, 4, 16);
      rs += __shfl_xor(rs, 2, 16);
      rs += __shfl_xor(rs, 1, 16);
      lrun[i] = lrun[i] * corr + rs;
      mrun[i] = mnew;
#pragma unroll
      for (int gI = 0; gI < 8; ++gI) oacc[gI][i] *= corr;
    }
    // ---- P: C-layout -> A-fragment layout via per-wave LDS transpose ----
#pragma unroll
    for (int i = 0; i < 8; ++i) {
      Psh[wave][(i + hb * 8) * 40 + lm]      = (_Float16)s0[i];
      Psh[wave][(i + hb * 8) * 40 + 16 + lm] = (_Float16)s1[i];
    }
    asm volatile("s_wait_dscnt 0" ::: "memory");
    const _Float16* pp = &Psh[wave][lm * 40 + koff];
    v16h pf = ld_frag2(pp, pp + 8);
    // ---- O += P * V ----
#pragma unroll
    for (int gI = 0; gI < 8; ++gI) {
      const _Float16* vp = Vh + (size_t)(gI * 16 + lm) * Lp + j0 + hb * 16;
      v16h bv = ld_frag2(vp, vp + 8);
      oacc[gI] = WMMA_F16(pf, bv, oacc[gI]);
    }
  }
  // ---- epilogue: O / l, write f16 [q][h*128+d] ----
#pragma unroll
  for (int i = 0; i < 8; ++i) {
    float inv = 1.0f / lrun[i];
    size_t base = (size_t)(q0 + i + hb * 8) * DIM + h * HD;
#pragma unroll
    for (int gI = 0; gI < 8; ++gI)
      O[base + gI * 16 + lm] = (_Float16)(oacc[gI][i] * inv);
  }
}

// ---------------------------------------------------------------------------
extern "C" void kernel_launch(void* const* d_in, const int* in_sizes, int n_in,
                              void* d_out, int out_size, void* d_ws, size_t ws_size,
                              hipStream_t stream) {
  const float* x          = (const float*)d_in[0];
  const int*   seq_lens   = (const int*)d_in[1];
  const int*   grid_sizes = (const int*)d_in[2];
  const float* freqs      = (const float*)d_in[3];
  const float* Wq = (const float*)d_in[4];
  const float* bq = (const float*)d_in[5];
  const float* Wk = (const float*)d_in[6];
  const float* bk = (const float*)d_in[7];
  const float* Wv = (const float*)d_in[8];
  const float* bv = (const float*)d_in[9];
  const float* Wo = (const float*)d_in[10];
  const float* bo = (const float*)d_in[11];
  const float* gq = (const float*)d_in[12];
  const float* gk = (const float*)d_in[13];

  char* ws = (char*)d_ws;
  size_t off = 0;
  auto alloc = [&](size_t bytes) -> void* {
    void* p = ws + off;
    off += (bytes + 255) & ~(size_t)255;
    return p;
  };
  _Float16* Xh  = (_Float16*)alloc((size_t)LPAD * DIM * 2);
  _Float16* Wqt = (_Float16*)alloc((size_t)DIM * DIM * 2);   // transposed f16 weights
  _Float16* Wkt = (_Float16*)alloc((size_t)DIM * DIM * 2);
  _Float16* Wvt = (_Float16*)alloc((size_t)DIM * DIM * 2);
  _Float16* Wot = (_Float16*)alloc((size_t)DIM * DIM * 2);
  _Float16* Qg  = (_Float16*)alloc((size_t)LPAD * DIM * 2);
  _Float16* Kg  = (_Float16*)alloc((size_t)LPAD * DIM * 2);
  _Float16* Vg  = (_Float16*)alloc((size_t)LPAD * DIM * 2);
  _Float16* Qr  = (_Float16*)alloc((size_t)NH * LPAD * HD * 2);
  _Float16* Kr  = (_Float16*)alloc((size_t)NH * LPAD * HD * 2);
  _Float16* Vt  = (_Float16*)alloc((size_t)NH * HD * LPAD * 2);
  _Float16* AttO = Qg;  // Qg fully consumed before attention writes

  // 1) casts (+ weight transpose)
  cast_pad_kernel<<<2048, 256, 0, stream>>>(x, Xh, LSEQ, DIM, LPAD);
  cast_transpose_kernel<<<1024, 256, 0, stream>>>(Wq, Wqt, DIM, DIM);
  cast_transpose_kernel<<<1024, 256, 0, stream>>>(Wk, Wkt, DIM, DIM);
  cast_transpose_kernel<<<1024, 256, 0, stream>>>(Wv, Wvt, DIM, DIM);
  cast_transpose_kernel<<<1024, 256, 0, stream>>>(Wo, Wot, DIM, DIM);

  // 2) QKV projections (f16 out)
  dim3 ggrid(LPAD / 128, DIM / 128);
  gemm_f16_kernel<<<ggrid, 256, 0, stream>>>(Xh, Wqt, bq, Qg, nullptr, LPAD, DIM, DIM, LPAD);
  gemm_f16_kernel<<<ggrid, 256, 0, stream>>>(Xh, Wkt, bk, Kg, nullptr, LPAD, DIM, DIM, LPAD);
  gemm_f16_kernel<<<ggrid, 256, 0, stream>>>(Xh, Wvt, bv, Vg, nullptr, LPAD, DIM, DIM, LPAD);

  // 3) norm + rope + repack
  rmsnorm_rope_kernel<<<LPAD, 256, 0, stream>>>(Qg, gq, freqs, grid_sizes, Qr, LSEQ, LPAD);
  rmsnorm_rope_kernel<<<LPAD, 256, 0, stream>>>(Kg, gk, freqs, grid_sizes, Kr, LSEQ, LPAD);
  vpack_kernel<<<LPAD, 256, 0, stream>>>(Vg, Vt, LSEQ, LPAD);

  // 4) flash attention (writes AttO = Qg region)
  attn_kernel<<<dim3(LPAD / 128, NH), 256, 0, stream>>>(Qr, Kr, Vt, AttO, seq_lens, LPAD);

  // 5) output projection (f32 out, rows < 5000)
  gemm_f16_kernel<<<ggrid, 256, 0, stream>>>(AttO, Wot, bo, nullptr, (float*)d_out,
                                             LPAD, DIM, DIM, LSEQ);
}
